// SelfAttention_71296457113769
// MI455X (gfx1250) — matrix operand order repaired and checked by reading, not compiled, not run on hardware
//
#include <hip/hip_runtime.h>
#include <stdint.h>

#define B_   8
#define C_   256
#define HW_  4096
#define G_   8
#define CPG_ 32
#define O3_  768
#define SROW 4100   // 4096 + 4 dwords pad: kills LDS bank conflicts on row-strided access

typedef __attribute__((ext_vector_type(16))) __bf16   v16bf;
typedef __attribute__((ext_vector_type(8)))  float    v8f;
typedef __attribute__((ext_vector_type(4)))  uint32_t v4u;
typedef __attribute__((ext_vector_type(4)))  float    v4f;

union ABRegs {
  uint32_t u[8];
  v4u      u4[2];
  v16bf    v;
};

__device__ __forceinline__ uint32_t f2bf(float f) {
  uint32_t u = __float_as_uint(f);
  return (u + 0x7FFFu + ((u >> 16) & 1u)) >> 16;   // round-to-nearest-even
}
__device__ __forceinline__ uint32_t pack2(float lo, float hi) {
  return f2bf(lo) | (f2bf(hi) << 16);
}

__device__ __forceinline__ v8f wmma_bf16(ABRegs a, ABRegs b, v8f c) {
  // D = A(16x32 bf16) * B(32x16 bf16) + C(16x16 f32)
  return __builtin_amdgcn_wmma_f32_16x16x32_bf16(false, a.v, false, b.v,
                                                 (short)0, c, false, false);
}

// ---------------------------------------------------------------------------
// Kernel 1: convert qkv_w [768x256] and proj_w [256x256] f32 -> bf16
// ---------------------------------------------------------------------------
__global__ void cvt_weights_kernel(const float* __restrict__ qw,
                                   const float* __restrict__ pw,
                                   uint16_t* __restrict__ qwb,
                                   uint16_t* __restrict__ pwb) {
  int i = blockIdx.x * 256 + threadIdx.x;
  if (i < O3_ * C_) qwb[i] = (uint16_t)f2bf(qw[i]);
  if (i < C_ * C_)  pwb[i] = (uint16_t)f2bf(pw[i]);
}

// ---------------------------------------------------------------------------
// Kernel 2: GroupNorm + affine + transpose to token-major bf16 hT[b][n][c]
// one block per (batch, group); 256 threads
// ---------------------------------------------------------------------------
__global__ void gn_transpose_kernel(const float* __restrict__ x,
                                    const float* __restrict__ w,
                                    const float* __restrict__ bias,
                                    uint16_t* __restrict__ hT) {
  const int b = blockIdx.x / G_;
  const int g = blockIdx.x % G_;
  const int tid = threadIdx.x;
  const float* xg = x + ((size_t)b * C_ + g * CPG_) * HW_;

  __shared__ float    red[256];
  __shared__ float    red2[256];
  __shared__ uint16_t tile[CPG_][64 + 2];

  float s = 0.f, s2 = 0.f;
  for (int i = tid; i < CPG_ * HW_; i += 256) {
    float v = xg[i]; s += v; s2 += v * v;
  }
  red[tid] = s; red2[tid] = s2;
  __syncthreads();
  for (int off = 128; off > 0; off >>= 1) {
    if (tid < off) { red[tid] += red[tid + off]; red2[tid] += red2[tid + off]; }
    __syncthreads();
  }
  const float inv_n = 1.0f / (float)(CPG_ * HW_);
  const float mean = red[0] * inv_n;
  const float var  = red2[0] * inv_n - mean * mean;
  const float rstd = rsqrtf(var + 1e-5f);
  __syncthreads();

  for (int n0 = 0; n0 < HW_; n0 += 64) {
    for (int i = tid; i < CPG_ * 64; i += 256) {
      int c = i >> 6, n = i & 63;
      float gm = w[g * CPG_ + c], bt = bias[g * CPG_ + c];
      float v = (xg[(size_t)c * HW_ + n0 + n] - mean) * rstd * gm + bt;
      tile[c][n] = (uint16_t)f2bf(v);
    }
    __syncthreads();
    {
      int n  = tid >> 2;           // 0..63
      int cb = (tid & 3) * 8;      // 0,8,16,24
      v4u pk;
      #pragma unroll
      for (int j = 0; j < 4; j++)
        pk[j] = (uint32_t)tile[cb + 2 * j][n] | ((uint32_t)tile[cb + 2 * j + 1][n] << 16);
      uint16_t* dst = hT + ((size_t)b * HW_ + n0 + n) * C_ + g * CPG_ + cb;
      *(v4u*)dst = pk;
    }
    __syncthreads();
  }
}

// ---------------------------------------------------------------------------
// Kernel 3: QKV GEMM  qkv[o][n] = sum_c W[o][c] * hT[n][c] + b[o]
// one wave per 16x16 tile; Q/K stored token-major bf16, V channel-major bf16
// ---------------------------------------------------------------------------
__global__ void qkv_gemm_kernel(const uint16_t* __restrict__ Wq,
                                const float* __restrict__ bq,
                                const uint16_t* __restrict__ hT,
                                uint16_t* __restrict__ qws,
                                uint16_t* __restrict__ kws,
                                uint16_t* __restrict__ vws) {
  const int wave = threadIdx.x >> 5;
  const int lane = threadIdx.x & 31;
  const int tile = blockIdx.x * 4 + wave;      // 8*48*256 tiles total
  const int nt = tile & 255;
  const int rest = tile >> 8;
  const int ot = rest % 48;
  const int b  = rest / 48;
  const int o0 = ot * 16, n0 = nt * 16;
  const int ll = lane & 15, hi = lane >> 4;

  const uint16_t* hb = hT + (size_t)b * HW_ * C_;
  v8f acc = {};
  #pragma unroll
  for (int k0 = 0; k0 < C_; k0 += 32) {
    ABRegs a, bm;
    // A (16x32): lanes0-15 K{0..7,16..23}, lanes16-31 K{8..15,24..31}
    const uint16_t* ap = Wq + (size_t)(o0 + ll) * C_ + k0 + hi * 8;
    a.u4[0] = *(const v4u*)ap;
    a.u4[1] = *(const v4u*)(ap + 16);
    // B (32x16): lanes0-15 K{0..7,16..23}, lanes16-31 K{8..15,24..31}, col = lane&15
    const uint16_t* bp = hb + (size_t)(n0 + ll) * C_ + k0 + hi * 8;
    bm.u4[0] = *(const v4u*)bp;
    bm.u4[1] = *(const v4u*)(bp + 16);
    acc = wmma_bf16(a, bm, acc);
  }
  const int orow = o0 + hi * 8;
  float vals[8];
  #pragma unroll
  for (int r = 0; r < 8; r++) vals[r] = acc[r] + bq[orow + r];

  if (o0 < C_) {                      // Q: token-major, fold scale C^-0.5 = 1/16
    v4u pk;
    #pragma unroll
    for (int j = 0; j < 4; j++)
      pk[j] = pack2(vals[2 * j] * 0.0625f, vals[2 * j + 1] * 0.0625f);
    uint16_t* dst = qws + ((size_t)b * HW_ + n0 + ll) * C_ + orow;
    *(v4u*)dst = pk;
  } else if (o0 < 2 * C_) {           // K: token-major
    v4u pk;
    #pragma unroll
    for (int j = 0; j < 4; j++)
      pk[j] = pack2(vals[2 * j], vals[2 * j + 1]);
    uint16_t* dst = kws + ((size_t)b * HW_ + n0 + ll) * C_ + (orow - C_);
    *(v4u*)dst = pk;
  } else {                            // V: channel-major
    #pragma unroll
    for (int r = 0; r < 8; r++) {
      int c = orow - 2 * C_ + r;
      vws[((size_t)b * C_ + c) * HW_ + n0 + ll] = (uint16_t)f2bf(vals[r]);
    }
  }
}

// ---------------------------------------------------------------------------
// Kernel 4: attention. One 8-wave WG per (batch, 16-query tile).
// S[16][4096] f32 lives entirely in LDS (257 KB of the 320 KB WGP LDS).
// Softmax normalization packs P to bf16 IN PLACE over the f32 buffer so the
// P.V phase reads ready-made A-operand bf16 directly from LDS.
// ---------------------------------------------------------------------------
__global__ void attn_kernel(const uint16_t* __restrict__ q,
                            const uint16_t* __restrict__ k,
                            const uint16_t* __restrict__ v,
                            uint16_t* __restrict__ ao) {
  extern __shared__ float smem[];
  float* S   = smem;               // [16][SROW]
  float* red = smem + 16 * SROW;   // [256]

  const int b  = blockIdx.x >> 8;
  const int qt = blockIdx.x & 255;
  const int n0 = qt * 16;
  const int tid  = threadIdx.x;
  const int wave = tid >> 5;
  const int lane = tid & 31;
  const int ll = lane & 15, hi = lane >> 4;

  const uint16_t* qb = q + (size_t)b * HW_ * C_;
  const uint16_t* kb = k + (size_t)b * HW_ * C_;
  const uint16_t* vb = v + (size_t)b * C_ * HW_;

  // Q tile resident in A-layout regs: 8 chunks x 8 VGPRs
  ABRegs qa[8];
  #pragma unroll
  for (int c8 = 0; c8 < 8; c8++) {
    const uint16_t* ap = qb + (size_t)(n0 + ll) * C_ + c8 * 32 + hi * 8;
    qa[c8].u4[0] = *(const v4u*)ap;
    qa[c8].u4[1] = *(const v4u*)(ap + 16);
  }

  // Phase 1: S = Q K^T  (each wave strides over key tiles)
  for (int mt = wave; mt < 256; mt += 8) {
    const int m0 = mt * 16;
    v8f acc = {};
    #pragma unroll
    for (int c8 = 0; c8 < 8; c8++) {
      ABRegs bm;
      const uint16_t* bp = kb + (size_t)(m0 + ll) * C_ + c8 * 32 + hi * 8;
      bm.u4[0] = *(const v4u*)bp;
      bm.u4[1] = *(const v4u*)(bp + 16);
      acc = wmma_bf16(qa[c8], bm, acc);
    }
    #pragma unroll
    for (int r = 0; r < 8; r++)
      S[(size_t)(hi * 8 + r) * SROW + m0 + ll] = acc[r];
  }
  __syncthreads();

  // Phase 2: row softmax over 4096 keys (16 threads per row, strided =>
  // conflict-free banks), then normalize + pack P to bf16 in place.
  {
    const int r = tid >> 4;
    const int c = tid & 15;
    float* Sr = S + (size_t)r * SROW;
    float mx = -3.0e38f;
    for (int m = c; m < HW_; m += 16) mx = fmaxf(mx, Sr[m]);
    red[tid] = mx; __syncthreads();
    for (int off = 8; off > 0; off >>= 1) {
      if (c < off) red[tid] = fmaxf(red[tid], red[tid + off]);
      __syncthreads();
    }
    mx = red[r * 16];
    __syncthreads();
    float sum = 0.f;
    for (int m = c; m < HW_; m += 16) {
      float e = __expf(Sr[m] - mx);
      Sr[m] = e; sum += e;
    }
    red[tid] = sum; __syncthreads();
    for (int off = 8; off > 0; off >>= 1) {
      if (c < off) red[tid] += red[tid + off];
      __syncthreads();
    }
    const float inv = 1.0f / red[r * 16];

    // In-place normalize + pack: bf16 element m lands at uint16 offset m of
    // the row (dword m/2), i.e. over the low half of the f32 row. Chunk j:
    // reads cover floats [128j, 128j+128), writes cover floats [64j, 64j+64)
    // -> with per-chunk lockstep (one barrier between reads and writes) no
    // packed write ever clobbers an unread f32 score.
    uint32_t* Pr32 = (uint32_t*)Sr;
    for (int jc = 0; jc < 32; jc++) {
      float f[8];
      #pragma unroll
      for (int t = 0; t < 4; t++) {
        int m = 2 * c + 32 * (4 * jc + t);
        f[2 * t]     = Sr[m];
        f[2 * t + 1] = Sr[m + 1];
      }
      __syncthreads();
      #pragma unroll
      for (int t = 0; t < 4; t++) {
        int m = 2 * c + 32 * (4 * jc + t);
        Pr32[m >> 1] = pack2(f[2 * t] * inv, f[2 * t + 1] * inv);
      }
    }
  }
  __syncthreads();

  // Phase 3: out = P V ; wave owns channels [wave*32, wave*32+32).
  // A-operand (P) read directly as bf16 from LDS (2 x ds_load_b128, no VALU
  // repack), shared across both 16-wide channel tiles.
  const int c0 = wave * 32;
  v8f acc0 = {}, acc1 = {};
  const uint16_t* Pb = (const uint16_t*)S;
  for (int mb = 0; mb < HW_; mb += 32) {
    ABRegs pa, vm0, vm1;
    // lane row = ll; lo lanes K{0..7,16..23}, hi lanes K{8..15,24..31}
    const uint16_t* prow = Pb + (size_t)ll * (2 * SROW) + mb + hi * 8;
    pa.u4[0] = *(const v4u*)prow;
    pa.u4[1] = *(const v4u*)(prow + 16);
    // V B-layout: col = channel, K = keys contiguous (channel-major V)
    const uint16_t* vp0 = vb + (size_t)(c0 + ll) * HW_ + mb + hi * 8;
    const uint16_t* vp1 = vp0 + (size_t)16 * HW_;
    vm0.u4[0] = *(const v4u*)vp0;
    vm0.u4[1] = *(const v4u*)(vp0 + 16);
    vm1.u4[0] = *(const v4u*)vp1;
    vm1.u4[1] = *(const v4u*)(vp1 + 16);
    acc0 = wmma_bf16(pa, vm0, acc0);
    acc1 = wmma_bf16(pa, vm1, acc1);
  }
  uint16_t* aop = ao + ((size_t)b * HW_ + n0) * C_;
  #pragma unroll
  for (int r = 0; r < 8; r++) {
    aop[(size_t)(hi * 8 + r) * C_ + c0 + ll]      = (uint16_t)f2bf(acc0[r]);
    aop[(size_t)(hi * 8 + r) * C_ + c0 + 16 + ll] = (uint16_t)f2bf(acc1[r]);
  }
}

// ---------------------------------------------------------------------------
// Kernel 5: proj GEMM + bias + residual, f32 output in [b][c][h][w]
// ---------------------------------------------------------------------------
__global__ void proj_kernel(const uint16_t* __restrict__ Pw,
                            const float* __restrict__ pb,
                            const uint16_t* __restrict__ ao,
                            const float* __restrict__ x,
                            float* __restrict__ out) {
  const int wave = threadIdx.x >> 5;
  const int lane = threadIdx.x & 31;
  const int tile = blockIdx.x * 4 + wave;      // 8*16*256 tiles
  const int nt = tile & 255;
  const int rest = tile >> 8;
  const int ot = rest & 15;
  const int b  = rest >> 4;
  const int o0 = ot * 16, n0 = nt * 16;
  const int ll = lane & 15, hi = lane >> 4;

  const uint16_t* ab = ao + (size_t)b * HW_ * C_;
  v8f acc = {};
  #pragma unroll
  for (int k0 = 0; k0 < C_; k0 += 32) {
    ABRegs a, bm;
    const uint16_t* ap = Pw + (size_t)(o0 + ll) * C_ + k0 + hi * 8;
    a.u4[0] = *(const v4u*)ap;
    a.u4[1] = *(const v4u*)(ap + 16);
    const uint16_t* bp = ab + (size_t)(n0 + ll) * C_ + k0 + hi * 8;
    bm.u4[0] = *(const v4u*)bp;
    bm.u4[1] = *(const v4u*)(bp + 16);
    acc = wmma_bf16(a, bm, acc);
  }
  #pragma unroll
  for (int r = 0; r < 8; r++) {
    int o = o0 + hi * 8 + r;
    size_t idx = ((size_t)b * C_ + o) * HW_ + n0 + ll;
    out[idx] = acc[r] + pb[o] + x[idx];
  }
}

// ---------------------------------------------------------------------------
extern "C" void kernel_launch(void* const* d_in, const int* in_sizes, int n_in,
                              void* d_out, int out_size, void* d_ws, size_t ws_size,
                              hipStream_t stream) {
  (void)in_sizes; (void)n_in; (void)out_size; (void)ws_size;
  const float* x    = (const float*)d_in[0];
  const float* gnw  = (const float*)d_in[1];
  const float* gnb  = (const float*)d_in[2];
  const float* qkvw = (const float*)d_in[3];
  const float* qkvb = (const float*)d_in[4];
  const float* pw   = (const float*)d_in[5];
  const float* pb   = (const float*)d_in[6];
  float* out = (float*)d_out;

  char* ws = (char*)d_ws;
  size_t off = 0;
  const size_t tokCh = (size_t)B_ * HW_ * C_;             // 8 * 4096 * 256
  uint16_t* hT  = (uint16_t*)(ws + off); off += tokCh * 2;
  uint16_t* qws = (uint16_t*)(ws + off); off += tokCh * 2;
  uint16_t* kws = (uint16_t*)(ws + off); off += tokCh * 2;
  uint16_t* vws = (uint16_t*)(ws + off); off += tokCh * 2;
  uint16_t* aow = (uint16_t*)(ws + off); off += tokCh * 2;
  uint16_t* qwb = (uint16_t*)(ws + off); off += (size_t)O3_ * C_ * 2;
  uint16_t* pwb = (uint16_t*)(ws + off); off += (size_t)C_ * C_ * 2;

  cvt_weights_kernel<<<dim3(768), dim3(256), 0, stream>>>(qkvw, pw, qwb, pwb);
  gn_transpose_kernel<<<dim3(64), dim3(256), 0, stream>>>(x, gnw, gnb, hT);
  qkv_gemm_kernel<<<dim3(24576), dim3(128), 0, stream>>>(qwb, qkvb, hT, qws, kws, vws);
  const size_t smem = (16 * (size_t)SROW + 256) * sizeof(float);  // 263,424 B < 320 KB
  attn_kernel<<<dim3(2048), dim3(256), smem, stream>>>(qws, kws, vws, aow);
  proj_kernel<<<dim3(8192), dim3(128), 0, stream>>>(pwb, pb, aow, x, out);
}
